// AxialAttention_850403524777
// MI455X (gfx1250) — compile-verified, matrix operand order checked by reading, not run on hardware
//
#include <hip/hip_runtime.h>
#include <math.h>

typedef __attribute__((ext_vector_type(16))) _Float16 v16h;
typedef __attribute__((ext_vector_type(8)))  _Float16 v8h;
typedef __attribute__((ext_vector_type(8)))  float    v8f;
typedef __attribute__((ext_vector_type(4)))  float    v4f;

#define BB 4
#define HH 256
#define WW 256
#define DD 128
#define TT 256            // line length
#define KV_STRIDE 132     // floats, padded (4-float skew -> conflict-free column access)
#define M_STRIDE  136     // halfs, padded (keeps 16B alignment of 32-half rows)
#define G_STRIDE  132     // floats

// ---------------- prep: M = Wq^T @ Wk  (per axis), stored f16 row-major ----------------
__global__ void axial_prepM(const float* __restrict__ Wq_h, const float* __restrict__ Wk_h,
                            const float* __restrict__ Wq_w, const float* __restrict__ Wk_w,
                            _Float16* __restrict__ Mout) {
    const int axis = blockIdx.y;           // 0 = h, 1 = w
    const int i = blockIdx.x;              // row of M (x-channel)
    const int j = threadIdx.x;             // col of M (kv-channel)
    const float* Wq = axis ? Wq_w : Wq_h;
    const float* Wk = axis ? Wk_w : Wk_h;
    float acc = 0.f;
#pragma unroll 8
    for (int o = 0; o < DD; ++o)
        acc += Wq[o * DD + i] * Wk[o * DD + j];   // M[i][j] = sum_o Wq[o][i]*Wk[o][j]
    Mout[(size_t)axis * DD * DD + i * DD + j] = (_Float16)acc;
}

// ---------------- main: one workgroup per line ----------------
__global__ __launch_bounds__(256, 1)
void axial_line(const float* __restrict__ x, const float* __restrict__ kv,
                const _Float16* __restrict__ Mf16, float* __restrict__ out,
                int axis, int accumulate) {
    __shared__ float    sKV[TT * KV_STRIDE];   // 135168 B, f32 line of kv
    __shared__ _Float16 sM [DD * M_STRIDE];    //  34816 B, M row-major f16
    __shared__ float    sG [TT * G_STRIDE];    // 135168 B, g = M . kv per token

    const int tid  = threadIdx.x;
    const int wv   = tid >> 5;
    const int lane = tid & 31;
    const int l16  = lane & 15;
    const int half = lane >> 4;
    const int bid  = blockIdx.x;               // 0..1023

    size_t tok0, tokStride;
    if (axis == 0) {                           // line along H at fixed (b, w)
        int b = bid >> 8, w = bid & 255;
        tok0 = ((size_t)b * HH * WW + w) * DD;
        tokStride = (size_t)WW * DD;
    } else {                                   // line along W at fixed (b, h)
        tok0 = (size_t)bid * WW * DD;
        tokStride = DD;
    }

    // ---- Phase 1: stage M (f16) and the kv line (f32) into LDS ----
    {
        const uint4* msrc = (const uint4*)Mf16;        // 2048 uint4 = 16384 halfs
#pragma unroll
        for (int q = tid; q < (DD * DD) / 8; q += 256) {
            int row = q >> 4, c8 = q & 15;             // 16 uint4 per 128-half row
            *(uint4*)&sM[row * M_STRIDE + c8 * 8] = msrc[q];
        }
        for (int r = wv; r < TT; r += 8) {
            v4f v = *(const v4f*)(kv + tok0 + (size_t)r * tokStride + lane * 4);
            *(v4f*)&sKV[r * KV_STRIDE + lane * 4] = v;
        }
    }
    __syncthreads();

    // ---- Phase 2: G[t][i] = sum_j kv[t][j] * M[i][j] via v_wmma_f32_16x16x32_f16 ----
    // wave wv owns token rows [wv*32, wv*32+32): M-tiles mt0=2*wv, mt1=2*wv+1
    {
        v16h A[8];                                     // [m2(2)][ks(4)]
#pragma unroll
        for (int m2 = 0; m2 < 2; ++m2) {
            int row = (wv * 2 + m2) * 16 + l16;        // lane holds A row l16 (both halves)
#pragma unroll
            for (int ks = 0; ks < 4; ++ks) {
                // ISA A-layout: lane(half,m) holds K = ks*32 + {half*8+0..7, 16+half*8+0..7}
                const float* p = &sKV[row * KV_STRIDE + ks * 32 + half * 8];
                v4f f0 = *(const v4f*)p, f1 = *(const v4f*)(p + 4);
                v4f f2 = *(const v4f*)(p + 16), f3 = *(const v4f*)(p + 20);
                v16h a;
#pragma unroll
                for (int t = 0; t < 4; ++t) {
                    a[t]      = (_Float16)f0[t];
                    a[4 + t]  = (_Float16)f1[t];
                    a[8 + t]  = (_Float16)f2[t];
                    a[12 + t] = (_Float16)f3[t];
                }
                A[m2 * 4 + ks] = a;
            }
        }
#pragma unroll
        for (int n = 0; n < 8; ++n) {
            v8f acc0 = {}, acc1 = {};
#pragma unroll
            for (int ks = 0; ks < 4; ++ks) {
                // B[k][n] = M[n][k]: lane(half,n16) holds col n, same K pattern, contiguous in M row
                int brow = n * 16 + l16;
                const _Float16* q = &sM[brow * M_STRIDE + ks * 32 + half * 8];
                v8h b0 = *(const v8h*)q, b1 = *(const v8h*)(q + 16);
                v16h bb;
#pragma unroll
                for (int t = 0; t < 8; ++t) { bb[t] = b0[t]; bb[8 + t] = b1[t]; }
                acc0 = __builtin_amdgcn_wmma_f32_16x16x32_f16(false, A[ks],     false, bb,
                                                              (short)0, acc0, false, false);
                acc1 = __builtin_amdgcn_wmma_f32_16x16x32_f16(false, A[4 + ks], false, bb,
                                                              (short)0, acc1, false, false);
            }
            int gcol = n * 16 + l16;
#pragma unroll
            for (int r = 0; r < 8; ++r) {              // C/D layout: vgpr r -> M = r + 8*half
                sG[(wv * 32 +      r + 8 * half) * G_STRIDE + gcol] = acc0[r];
                sG[(wv * 32 + 16 + r + 8 * half) * G_STRIDE + gcol] = acc1[r];
            }
        }
    }
    __syncthreads();

    // ---- Phase 3: banded dots + softmax + weighted kv sum ----
    const float SCALE = 0.08838834764831845f;          // 1/sqrt(128)
    const v4f zero4 = {0.f, 0.f, 0.f, 0.f};
    for (int rr = 0; rr < 32; ++rr) {
        const int i = wv * 32 + rr;                    // wave processes row i together
        const int c = lane * 4;                        // 4 channels per lane
        v4f xv = *(const v4f*)(x + tok0 + (size_t)i * tokStride + c);
        v4f g0 = *(const v4f*)&sG[i * G_STRIDE + c];
        v4f gm = (i > 0)      ? *(const v4f*)&sG[(i - 1) * G_STRIDE + c] : zero4;
        v4f gp = (i < TT - 1) ? *(const v4f*)&sG[(i + 1) * G_STRIDE + c] : zero4;
        float pm = xv[0]*gm[0] + xv[1]*gm[1] + xv[2]*gm[2] + xv[3]*gm[3];
        float p0 = xv[0]*g0[0] + xv[1]*g0[1] + xv[2]*g0[2] + xv[3]*g0[3];
        float pp = xv[0]*gp[0] + xv[1]*gp[1] + xv[2]*gp[2] + xv[3]*gp[3];
#pragma unroll
        for (int off = 16; off > 0; off >>= 1) {       // wave32 butterfly: all lanes get sums
            pm += __shfl_xor(pm, off, 32);
            p0 += __shfl_xor(p0, off, 32);
            pp += __shfl_xor(pp, off, 32);
        }
        float sm = (i > 0)      ? pm * SCALE : -__builtin_inff();
        float sp = (i < TT - 1) ? pp * SCALE : -__builtin_inff();
        float s0 = p0 * SCALE;
        float mx = fmaxf(s0, fmaxf(sm, sp));
        float em = expf(sm - mx), e0 = expf(s0 - mx), ep = expf(sp - mx);
        float inv = 1.0f / (em + e0 + ep);
        em *= inv; e0 *= inv; ep *= inv;

        v4f kv0 = *(const v4f*)&sKV[i * KV_STRIDE + c];         // full-precision kv
        v4f kvm = (i > 0)      ? *(const v4f*)&sKV[(i - 1) * KV_STRIDE + c] : zero4;
        v4f kvp = (i < TT - 1) ? *(const v4f*)&sKV[(i + 1) * KV_STRIDE + c] : zero4;
        v4f o = kv0 * e0 + kvm * em + kvp * ep;

        float* dst = out + tok0 + (size_t)i * tokStride + c;
        if (accumulate) o += *(const v4f*)dst;
        *(v4f*)dst = o;
    }
}

extern "C" void kernel_launch(void* const* d_in, const int* in_sizes, int n_in,
                              void* d_out, int out_size, void* d_ws, size_t ws_size,
                              hipStream_t stream) {
    (void)in_sizes; (void)n_in; (void)out_size; (void)ws_size;
    const float* x    = (const float*)d_in[0];
    const float* kv   = (const float*)d_in[1];
    const float* Wq_h = (const float*)d_in[2];
    const float* Wk_h = (const float*)d_in[3];
    const float* Wq_w = (const float*)d_in[4];
    const float* Wk_w = (const float*)d_in[5];
    float* out = (float*)d_out;
    _Float16* Mws = (_Float16*)d_ws;                   // 2 x 128x128 f16 = 64 KB

    axial_prepM<<<dim3(DD, 2), DD, 0, stream>>>(Wq_h, Wk_h, Wq_w, Wk_w, Mws);
    // axis 0 (height lines): writes out
    axial_line<<<BB * WW, 256, 0, stream>>>(x, kv, Mws,           out, 0, 0);
    // axis 1 (width lines): accumulates into out
    axial_line<<<BB * HH, 256, 0, stream>>>(x, kv, Mws + DD * DD, out, 1, 1);
}
